// FusedGPTQMLP_70205535421121
// MI455X (gfx1250) — compile-verified, matrix operand order checked by reading, not run on hardware
//
#include <hip/hip_runtime.h>
#include <hip/hip_fp16.h>

typedef __attribute__((ext_vector_type(16))) _Float16 v16h;
typedef __attribute__((ext_vector_type(8)))  float    v8f;

#define HID   4096
#define ITR   11008
#define LDA   72   // halfs; 144B row stride (16B aligned, odd multiple of 16B)
#define LDB   72
#define KB    64   // K-step (two WMMA K=32 chunks)

union H8 { uint4 u4; _Float16 h[8]; };

__device__ __forceinline__ unsigned lds_u32(const void* p) {
  // generic pointer into LDS: low 32 bits are the LDS byte offset
  return (unsigned)(unsigned long long)p;
}

__device__ __forceinline__ void async_b128(unsigned lds_byte, const void* gaddr) {
  asm volatile("global_load_async_to_lds_b128 %0, %1, off"
               :: "v"(lds_byte), "v"(gaddr) : "memory");
}

__device__ __forceinline__ void wait_async0() {
  asm volatile("s_wait_asynccnt 0x0" ::: "memory");
}

// Dequant one packed int32 (8 nibbles along K for one column n) to 8 fp16.
__device__ __forceinline__ uint4 dq8(int q32, float sf, float zf1) {
  H8 w;
#pragma unroll
  for (int j = 0; j < 8; ++j)
    w.h[j] = (_Float16)(((float)((q32 >> (4 * j)) & 15) - zf1) * sf);
  return w.u4;
}

// 16x32 fp16 fragment load per the CDNA5 16-bit A layout:
// lanes 0-15 hold K=0..7,16..23; lanes 16-31 hold K=8..15,24..31.
__device__ __forceinline__ v16h frag_ld(const _Float16* tile, int rowbase, int ld, int lane) {
  const int r  = rowbase + (lane & 15);
  const int kb = (lane >> 4) * 8;
  v16h f;
  *(uint4*)&f       = *(const uint4*)&tile[r * ld + kb];
  *((uint4*)&f + 1) = *(const uint4*)&tile[r * ld + 16 + kb];
  return f;
}

__device__ __forceinline__ float fast_sigmoid(float v) {
  return __builtin_amdgcn_rcpf(1.0f + __expf(-v));
}

__global__ __launch_bounds__(256)
void gptq_gateup_silu_wmma(const _Float16* __restrict__ x,
                           const int* __restrict__ gq, const _Float16* __restrict__ gs,
                           const int* __restrict__ gz,
                           const int* __restrict__ uq, const _Float16* __restrict__ us,
                           const int* __restrict__ uz,
                           _Float16* __restrict__ inter) {
  const int n0   = blockIdx.x * 64;
  const int m0   = blockIdx.y * 128;
  const int tid  = threadIdx.x;
  const int lane = tid & 31;
  const int wave = tid >> 5;
  const int wm   = (wave & 3) * 32;  // wave M offset within 128-row tile
  const int wn   = (wave >> 2) * 32; // wave N offset within 64-col tile

  __shared__ __align__(16) _Float16 Al[2][128 * LDA];
  __shared__ __align__(16) _Float16 Bg[2][64 * LDB];
  __shared__ __align__(16) _Float16 Bu[2][64 * LDB];

  v8f accg[2][2] = {};
  v8f accu[2][2] = {};

  // A staging: 128 x 64 halfs; thread -> 32 halfs (4x async b128)
  const int arow = tid >> 1;
  const int acol = (tid & 1) * 32;
  const unsigned a_lds[2] = { lds_u32(&Al[0][arow * LDA + acol]),
                              lds_u32(&Al[1][arow * LDA + acol]) };
  const _Float16* a_gp = x + (size_t)(m0 + arow) * HID + acol;

  // B staging: 8 qweight rows x 64 cols per matrix; thread -> rows {r, r+4}
  const int brow = tid >> 6;   // 0..3
  const int bcol = tid & 63;   // 0..63
  const int n    = n0 + bcol;
  const int zsh  = (n & 7) * 4;

  auto stage = [&](int k0, int b) {
    const unsigned al = a_lds[b];
#pragma unroll
    for (int c = 0; c < 4; ++c)
      async_b128(al + c * 16, a_gp + k0 + c * 8);

    const int g = k0 >> 7;  // k0 / GROUPSIZE
    const float sgf = (float)gs[(size_t)g * ITR + n];
    const float suf = (float)us[(size_t)g * ITR + n];
    const int   zg  = gz[(size_t)g * (ITR / 8) + (n >> 3)];
    const int   zu  = uz[(size_t)g * (ITR / 8) + (n >> 3)];
    const float zgf = (float)(((zg >> zsh) & 15) + 1);
    const float zuf = (float)(((zu >> zsh) & 15) + 1);
#pragma unroll
    for (int j = 0; j < 2; ++j) {
      const int r    = brow + j * 4;            // 0..7
      const int qrow = (k0 >> 3) + r;
      const int qg = gq[(size_t)qrow * ITR + n];
      const int qu = uq[(size_t)qrow * ITR + n];
      *(uint4*)&Bg[b][bcol * LDB + r * 8] = dq8(qg, sgf, zgf);
      *(uint4*)&Bu[b][bcol * LDB + r * 8] = dq8(qu, suf, zuf);
    }
  };

  constexpr int STEPS = HID / KB;
  stage(0, 0);
  for (int i = 0; i < STEPS; ++i) {
    wait_async0();
    __syncthreads();
    if (i + 1 < STEPS) stage((i + 1) * KB, (i + 1) & 1);

    const int b = i & 1;
#pragma unroll
    for (int c = 0; c < 2; ++c) {   // K chunk of 32 within the 64-step
      v16h af[2], bgf[2], buf[2];
#pragma unroll
      for (int t = 0; t < 2; ++t) {
        af[t]  = frag_ld(&Al[b][c * 32], wm + t * 16, LDA, lane);
        bgf[t] = frag_ld(&Bg[b][c * 32], wn + t * 16, LDB, lane);
        buf[t] = frag_ld(&Bu[b][c * 32], wn + t * 16, LDB, lane);
      }
#pragma unroll
      for (int tm = 0; tm < 2; ++tm)
#pragma unroll
        for (int tn = 0; tn < 2; ++tn) {
          accg[tm][tn] = __builtin_amdgcn_wmma_f32_16x16x32_f16(
              false, af[tm], false, bgf[tn], (short)0, accg[tm][tn], false, false);
          accu[tm][tn] = __builtin_amdgcn_wmma_f32_16x16x32_f16(
              false, af[tm], false, buf[tn], (short)0, accu[tm][tn], false, false);
        }
    }
  }

  // Epilogue: SiLU(gate)*up -> fp16. C layout: row = r + (lane>>4)*8, col = lane&15
  const int lcol  = lane & 15;
  const int rhalf = (lane >> 4) * 8;
#pragma unroll
  for (int tm = 0; tm < 2; ++tm)
#pragma unroll
    for (int tn = 0; tn < 2; ++tn)
#pragma unroll
      for (int r = 0; r < 8; ++r) {
        const float gv = accg[tm][tn][r];
        const float uv = accu[tm][tn][r];
        const float sv = gv * fast_sigmoid(gv) * uv;
        const int row = m0 + wm + tm * 16 + rhalf + r;
        const int col = n0 + wn + tn * 16 + lcol;
        inter[(size_t)row * ITR + col] = (_Float16)sv;
      }
}

__global__ __launch_bounds__(256)
void gptq_down_wmma(const _Float16* __restrict__ inter,
                    const int* __restrict__ dq_, const _Float16* __restrict__ dsc,
                    const int* __restrict__ dz,
                    _Float16* __restrict__ out) {
  const int n0   = blockIdx.x * 64;
  const int m0   = blockIdx.y * 128;
  const int tid  = threadIdx.x;
  const int lane = tid & 31;
  const int wave = tid >> 5;
  const int wm   = (wave & 3) * 32;
  const int wn   = (wave >> 2) * 32;

  __shared__ __align__(16) _Float16 Al[2][128 * LDA];
  __shared__ __align__(16) _Float16 Bd[2][64 * LDB];

  v8f acc[2][2] = {};

  const int arow = tid >> 1;
  const int acol = (tid & 1) * 32;
  const unsigned a_lds[2] = { lds_u32(&Al[0][arow * LDA + acol]),
                              lds_u32(&Al[1][arow * LDA + acol]) };
  const _Float16* a_gp = inter + (size_t)(m0 + arow) * ITR + acol;

  const int brow = tid >> 6;
  const int bcol = tid & 63;
  const int n    = n0 + bcol;
  const int zsh  = (n & 7) * 4;

  auto stage = [&](int k0, int b) {
    const unsigned al = a_lds[b];
#pragma unroll
    for (int c = 0; c < 4; ++c)
      async_b128(al + c * 16, a_gp + k0 + c * 8);

    const int g = k0 >> 7;
    const float sf  = (float)dsc[(size_t)g * HID + n];
    const int   z32 = dz[(size_t)g * (HID / 8) + (n >> 3)];
    const float zf1 = (float)(((z32 >> zsh) & 15) + 1);
#pragma unroll
    for (int j = 0; j < 2; ++j) {
      const int r    = brow + j * 4;
      const int qrow = (k0 >> 3) + r;
      const int q32  = dq_[(size_t)qrow * HID + n];
      *(uint4*)&Bd[b][bcol * LDB + r * 8] = dq8(q32, sf, zf1);
    }
  };

  constexpr int STEPS = ITR / KB;
  stage(0, 0);
  for (int i = 0; i < STEPS; ++i) {
    wait_async0();
    __syncthreads();
    if (i + 1 < STEPS) stage((i + 1) * KB, (i + 1) & 1);

    const int b = i & 1;
#pragma unroll
    for (int c = 0; c < 2; ++c) {
      v16h af[2], bf[2];
#pragma unroll
      for (int t = 0; t < 2; ++t) {
        af[t] = frag_ld(&Al[b][c * 32], wm + t * 16, LDA, lane);
        bf[t] = frag_ld(&Bd[b][c * 32], wn + t * 16, LDB, lane);
      }
#pragma unroll
      for (int tm = 0; tm < 2; ++tm)
#pragma unroll
        for (int tn = 0; tn < 2; ++tn)
          acc[tm][tn] = __builtin_amdgcn_wmma_f32_16x16x32_f16(
              false, af[tm], false, bf[tn], (short)0, acc[tm][tn], false, false);
    }
  }

  const int lcol  = lane & 15;
  const int rhalf = (lane >> 4) * 8;
#pragma unroll
  for (int tm = 0; tm < 2; ++tm)
#pragma unroll
    for (int tn = 0; tn < 2; ++tn)
#pragma unroll
      for (int r = 0; r < 8; ++r) {
        const int row = m0 + wm + tm * 16 + rhalf + r;
        const int col = n0 + wn + tn * 16 + lcol;
        out[(size_t)row * HID + col] = (_Float16)acc[tm][tn][r];
      }
}

extern "C" void kernel_launch(void* const* d_in, const int* in_sizes, int n_in,
                              void* d_out, int out_size, void* d_ws, size_t ws_size,
                              hipStream_t stream) {
  const _Float16* x  = (const _Float16*)d_in[0];
  const int* gq      = (const int*)d_in[1];
  const _Float16* gs = (const _Float16*)d_in[2];
  const int* gz      = (const int*)d_in[3];
  const int* uq      = (const int*)d_in[4];
  const _Float16* us = (const _Float16*)d_in[5];
  const int* uz      = (const int*)d_in[6];
  const int* dq      = (const int*)d_in[7];
  const _Float16* ds = (const _Float16*)d_in[8];
  const int* dz      = (const int*)d_in[9];

  _Float16* out   = (_Float16*)d_out;
  _Float16* inter = (_Float16*)d_ws;   // M x ITR fp16 (~86 MB)

  const int M = in_sizes[0] / HID;     // 4096

  gptq_gateup_silu_wmma<<<dim3(ITR / 64, M / 128), dim3(256), 0, stream>>>(
      x, gq, gs, gz, uq, us, uz, inter);
  gptq_down_wmma<<<dim3(HID / 64, M / 128), dim3(256), 0, stream>>>(
      inter, dq, ds, dz, out);
}
